// MoeMamba_7679401525693
// MI455X (gfx1250) — matrix operand order repaired and checked by reading, not compiled, not run on hardware
//
#include <hip/hip_runtime.h>
#include <hip/hip_bf16.h>

#define T_  2
#define B_  8
#define C_  256
#define D_  64
#define E_  15
#define R_  64
#define HW_ 3136
#define K_  9

typedef __attribute__((ext_vector_type(16))) _Float16 v16h;
typedef __attribute__((ext_vector_type(8)))  _Float16 v8h;
typedef __attribute__((ext_vector_type(8)))  float    v8f;

__device__ __forceinline__ float gelu_exact(float v) {
    return 0.5f * v * (1.0f + erff(v * 0.70710678118654752440f));
}

// ---------------------------------------------------------------------------
// Kernel 1: feat = GELU(LN(conv1(x))); partial spatial sums of feat per block.
// grid (T*B, 14), block 256 = 4 pixels x 64 channels. No atomics (determinism).
// ---------------------------------------------------------------------------
__global__ __launch_bounds__(256)
void k_router_feat(const float* __restrict__ x, const float* __restrict__ w1,
                   const float* __restrict__ b1, const float* __restrict__ lng,
                   const float* __restrict__ lnb, float* __restrict__ fmp) {
    const int tb = blockIdx.x, t = tb / B_;
    const int tid = threadIdx.x;
    const int p = tid >> 6, d = tid & 63;
    __shared__ float xp[4][C_];
    __shared__ float yb[4][D_];
    __shared__ float fred[256];
    const float* xtb = x + (size_t)tb * C_ * HW_;
    const float* wrow = w1 + ((size_t)t * D_ + d) * C_;
    const float bia = b1[t * D_ + d];
    const float g   = lng[t * D_ + d];
    const float be  = lnb[t * D_ + d];
    float acc = 0.0f;
    const int sbeg = blockIdx.y * 224;
    for (int s0 = sbeg; s0 < sbeg + 224; s0 += 4) {
        float4 xv = *(const float4*)(xtb + (size_t)tid * HW_ + s0);
        xp[0][tid] = xv.x; xp[1][tid] = xv.y; xp[2][tid] = xv.z; xp[3][tid] = xv.w;
        __syncthreads();
        float y = bia;
        for (int c = 0; c < C_; ++c) y += xp[p][c] * wrow[c];
        yb[p][d] = y;
        __syncthreads();
        float m = 0.0f;
        for (int j = 0; j < D_; ++j) m += yb[p][j];
        m *= (1.0f / D_);
        float v = 0.0f;
        for (int j = 0; j < D_; ++j) { float dd = yb[p][j] - m; v += dd * dd; }
        v *= (1.0f / D_);
        acc += gelu_exact((y - m) * rsqrtf(v + 1e-5f) * g + be);
        __syncthreads();
    }
    fred[tid] = acc;
    __syncthreads();
    if (p == 0)
        fmp[((size_t)tb * 14 + blockIdx.y) * D_ + d] =
            fred[d] + fred[64 + d] + fred[128 + d] + fred[192 + d];
}

// ---------------------------------------------------------------------------
// Kernel 2: fs[t,b,c] = sum_s x[t,b,c,s] * r_sp_w[t,s].  grid T*B*C, block 256.
// ---------------------------------------------------------------------------
__global__ __launch_bounds__(256)
void k_router_sp(const float* __restrict__ x, const float* __restrict__ spw,
                 float* __restrict__ fs) {
    const int idx = blockIdx.x;                 // tb*256 + c
    const int t = (idx >> 8) / B_;
    const int tid = threadIdx.x;
    const float* xr = x + (size_t)idx * HW_;
    const float* wr = spw + (size_t)t * HW_;
    float a = 0.0f;
    for (int s = tid; s < HW_; s += 256) a += xr[s] * wr[s];
    __shared__ float red[256];
    red[tid] = a;
    __syncthreads();
    for (int st = 128; st > 0; st >>= 1) {
        if (tid < st) red[tid] += red[tid + st];
        __syncthreads();
    }
    if (tid == 0) fs[idx] = red[0];
}

// ---------------------------------------------------------------------------
// Kernel 3: gate — LN/GELU on fs, conv3 on fm, conv2, logits, softmax, top-9.
// grid T*B, block 256.
// ---------------------------------------------------------------------------
__global__ __launch_bounds__(256)
void k_router_gate(const float* __restrict__ fmp, const float* __restrict__ fs,
                   const float* __restrict__ spb, const float* __restrict__ lnspg,
                   const float* __restrict__ lnspb,
                   const float* __restrict__ w3, const float* __restrict__ b3,
                   const float* __restrict__ w2, const float* __restrict__ b2,
                   const float* __restrict__ wout, const float* __restrict__ bout,
                   float* __restrict__ route1, float* __restrict__ prob) {
    const int tb = blockIdx.x, t = tb / B_;
    const int tid = threadIdx.x;
    __shared__ float fsv[C_], fsg[C_], fmv[D_], cat[2 * D_], lg[2 * E_ + 1];
    fsv[tid] = fs[tb * C_ + tid] + spb[t];
    if (tid < D_) {
        float s = 0.0f;
        for (int yb = 0; yb < 14; ++yb) s += fmp[((size_t)tb * 14 + yb) * D_ + tid];
        fmv[tid] = s * (1.0f / HW_);
    }
    __syncthreads();
    float m = 0.0f;
    for (int c = 0; c < C_; ++c) m += fsv[c];
    m *= (1.0f / C_);
    float v = 0.0f;
    for (int c = 0; c < C_; ++c) { float dd = fsv[c] - m; v += dd * dd; }
    v *= (1.0f / C_);
    fsg[tid] = gelu_exact((fsv[tid] - m) * rsqrtf(v + 1e-5f) * lnspg[t * C_ + tid]
                          + lnspb[t * C_ + tid]);
    __syncthreads();
    if (tid < D_) {
        float a = b3[t * D_ + tid];
        const float* w = w3 + ((size_t)t * D_ + tid) * D_;
        for (int dd = 0; dd < D_; ++dd) a += fmv[dd] * w[dd];
        cat[tid] = a;
    } else if (tid < 2 * D_) {
        const int dd = tid - D_;
        float a = b2[t * D_ + dd];
        const float* w = w2 + ((size_t)t * D_ + dd) * C_;
        for (int c = 0; c < C_; ++c) a += fsg[c] * w[c];
        cat[tid] = a;
    }
    __syncthreads();
    if (tid < 2 * E_ + 1) {
        float a = bout[t * (2 * E_ + 1) + tid];
        const float* w = wout + ((size_t)t * (2 * E_ + 1) + tid) * (2 * D_);
        for (int f = 0; f < 2 * D_; ++f) a += cat[f] * w[f];
        lg[tid] = a;
    }
    __syncthreads();
    if (tid == 0) {
        float mx = lg[0];
        for (int e = 1; e < E_; ++e) mx = fmaxf(mx, lg[e]);
        float pr[E_], sm = 0.0f;
        for (int e = 0; e < E_; ++e) { pr[e] = expf(lg[e] - mx); sm += pr[e]; }
        const float rs = 1.0f / sm;
        for (int e = 0; e < E_; ++e) pr[e] *= rs;
        float srt[E_];
        for (int e = 0; e < E_; ++e) srt[e] = pr[e];
        for (int i = 1; i < E_; ++i) {            // insertion sort, descending
            float key = srt[i]; int j = i - 1;
            while (j >= 0 && srt[j] < key) { srt[j + 1] = srt[j]; --j; }
            srt[j + 1] = key;
        }
        const float kth = srt[K_ - 1];
        for (int e = 0; e < E_; ++e)
            route1[tb * E_ + e] = (pr[e] >= kth) ? pr[e] : 0.0f;
        prob[tb] = lg[2 * E_];
    }
}

// ---------------------------------------------------------------------------
// Kernel 4: rank expansion  ME[e] = M_e @ W_e  (+bias),  PT[t] = pt_M @ pt_W.
// grid (C, E+T), block 256.
// ---------------------------------------------------------------------------
__global__ __launch_bounds__(256)
void k_rank_expand(const float* __restrict__ eW, const float* __restrict__ eWb,
                   const float* __restrict__ eM, const float* __restrict__ eMb,
                   const float* __restrict__ ptW, const float* __restrict__ ptWb,
                   const float* __restrict__ ptM, const float* __restrict__ ptMb,
                   float* __restrict__ ME, float* __restrict__ MEb,
                   float* __restrict__ PT, float* __restrict__ PTb) {
    const int c = blockIdx.x, m = blockIdx.y, tid = threadIdx.x;
    const float *Mm, *Wm, *Wbv, *Mbv;
    float *Od, *Ob;
    if (m < E_) {
        Mm = eM + ((size_t)m * C_ + c) * R_;
        Wm = eW + (size_t)m * R_ * C_;
        Wbv = eWb + m * R_; Mbv = eMb + (size_t)m * C_;
        Od = ME + ((size_t)m * C_ + c) * C_;   Ob = MEb + (size_t)m * C_;
    } else {
        const int t = m - E_;
        Mm = ptM + ((size_t)t * C_ + c) * R_;
        Wm = ptW + (size_t)t * R_ * C_;
        Wbv = ptWb + t * R_; Mbv = ptMb + (size_t)t * C_;
        Od = PT + ((size_t)t * C_ + c) * C_;   Ob = PTb + (size_t)t * C_;
    }
    __shared__ float Mrow[R_];
    if (tid < R_) Mrow[tid] = Mm[tid];
    __syncthreads();
    float a = 0.0f;
    for (int r = 0; r < R_; ++r) a += Mrow[r] * Wm[(size_t)r * C_ + tid];
    Od[tid] = a;
    if (tid == 0) {
        float bb = Mbv[c];
        for (int r = 0; r < R_; ++r) bb += Mrow[r] * Wbv[r];
        Ob[c] = bb;
    }
}

// ---------------------------------------------------------------------------
// Kernel 5: combine everything into per-(t,b) f16 matrix A + fp32 bias.
// grid (C, T*B), block 256.
// ---------------------------------------------------------------------------
__global__ __launch_bounds__(256)
void k_combine(const float* __restrict__ ME, const float* __restrict__ MEb,
               const float* __restrict__ PT, const float* __restrict__ PTb,
               const float* __restrict__ pgw, const float* __restrict__ pgb,
               const float* __restrict__ g1, const float* __restrict__ b1v,
               const float* __restrict__ g2, const float* __restrict__ b2v,
               const float* __restrict__ route1, const float* __restrict__ prob,
               _Float16* __restrict__ Ah, float* __restrict__ biasTB) {
    const int c = blockIdx.x, tb = blockIdx.y, t = tb / B_, tid = threadIdx.x;
    __shared__ float wv[E_];
    if (tid < E_) wv[tid] = route1[tb * E_ + tid];
    __syncthreads();
    const float pm  = prob[tb];
    const float inv = rsqrtf(1.0f + 1e-5f);
    const float s1 = inv * g1[t * C_ + c];
    const float s2 = inv * g2[t * C_ + c];
    float a = 0.0f;
    for (int e = 0; e < E_; ++e) {
        const float w = wv[e];
        if (w != 0.0f) a += w * ME[((size_t)e * C_ + c) * C_ + tid];
    }
    const float val = s2 * (s1 * a + pm * PT[((size_t)t * C_ + c) * C_ + tid]
                            + pgw[(size_t)c * C_ + tid]);
    Ah[((size_t)tb * C_ + c) * C_ + tid] = (_Float16)val;
    if (tid == 0) {
        float ab = 0.0f;
        for (int e = 0; e < E_; ++e) {
            const float w = wv[e];
            if (w != 0.0f) ab += w * MEb[e * C_ + c];
        }
        biasTB[tb * C_ + c] = s2 * (s1 * ab + b1v[t * C_ + c] + pm * PTb[t * C_ + c]
                                    + pgb[c]) + b2v[t * C_ + c];
    }
}

// ---------------------------------------------------------------------------
// Kernel 6: transpose+convert  XhT[tb][s][c] (f16) <- x[tb][c][s] (f32).
// LDS-tiled 64x64 transpose so both global sides are coalesced b128.
// grid (49, 4, 16), block 256.
// ---------------------------------------------------------------------------
__global__ __launch_bounds__(256)
void k_cvt_transpose(const float* __restrict__ x, _Float16* __restrict__ XhT) {
    const int tb = blockIdx.z;
    const int s0 = blockIdx.x * 64, c0 = blockIdx.y * 64;
    const int tid = threadIdx.x;
    const int row = tid >> 2;          // 0..63
    const int seg = tid & 3;           // 0..3 : 16-element segment
    __shared__ _Float16 tile[64][72];  // [c][s], padded
    const float* xr = x + ((size_t)tb * C_ + c0 + row) * HW_ + s0 + seg * 16;
    #pragma unroll
    for (int q = 0; q < 4; ++q) {
        float4 v = *(const float4*)(xr + q * 4);
        const int sl = seg * 16 + q * 4;
        tile[row][sl + 0] = (_Float16)v.x;
        tile[row][sl + 1] = (_Float16)v.y;
        tile[row][sl + 2] = (_Float16)v.z;
        tile[row][sl + 3] = (_Float16)v.w;
    }
    __syncthreads();
    _Float16* orow = XhT + ((size_t)tb * HW_ + s0 + row) * C_ + c0 + seg * 16;
    #pragma unroll
    for (int q = 0; q < 2; ++q) {
        v8h h;
        #pragma unroll
        for (int i = 0; i < 8; ++i) h[i] = tile[seg * 16 + q * 8 + i][row];
        *(v8h*)(orow + q * 8) = h;
    }
}

// ---------------------------------------------------------------------------
// Kernel 7: batched GEMM with WMMA.  out[tb] = A[tb](256x256,f16) * X + bias.
// B operand read K-contiguous from XhT: 2 x b128 per fragment, no cvt in loop.
// grid (49, 2, 16), block 256 = 8 waves; wave -> 16(M) x 64(N) tile,
// K-loop 256 in steps of 32 -> 8 x 4 v_wmma_f32_16x16x32_f16 per wave.
// ---------------------------------------------------------------------------
__global__ __launch_bounds__(256)
void k_gemm_wmma(const _Float16* __restrict__ XhT, const _Float16* __restrict__ Ah,
                 const float* __restrict__ biasTB, float* __restrict__ out) {
    const int tb   = blockIdx.z;
    const int lane = threadIdx.x & 31;
    const int wave = threadIdx.x >> 5;
    const int Mbase = blockIdx.y * 128 + wave * 16;
    const int Nbase = blockIdx.x * 64;
    const int l15 = lane & 15;
    const bool hi = (lane & 16) != 0;
    const _Float16* Arow = Ah + ((size_t)tb * C_ + (Mbase + l15)) * C_;
    const _Float16* Brow = XhT + ((size_t)tb * HW_ + Nbase + l15) * C_;   // n = Nbase+l15
    v8f acc0 = {}, acc1 = {}, acc2 = {}, acc3 = {};
    for (int k = 0; k < C_; k += 32) {
        // ---- A fragment (16x32 f16, M x K, K-contiguous per lane half) ----
        const int ka = k + (hi ? 16 : 0);
        const v8h alo = *(const v8h*)(Arow + ka);        // K = ka .. ka+7
        const v8h ahg = *(const v8h*)(Arow + ka + 8);    // K = ka+8 .. ka+15
        v16h af;
        #pragma unroll
        for (int i = 0; i < 8; ++i) { af[i] = alo[i]; af[8 + i] = ahg[i]; }
        // ---- B fragments (32x16 f16, K x N): K-contiguous vector loads ----
        const int kb = k + (hi ? 16 : 0);
        const v16h bf0 = *(const v16h*)(Brow + 0 * 16 * C_ + kb);
        const v16h bf1 = *(const v16h*)(Brow + 1 * 16 * C_ + kb);
        const v16h bf2 = *(const v16h*)(Brow + 2 * 16 * C_ + kb);
        const v16h bf3 = *(const v16h*)(Brow + 3 * 16 * C_ + kb);
        __builtin_prefetch(Brow + kb + 32, 0, 1);        // global_prefetch_b8
        acc0 = __builtin_amdgcn_wmma_f32_16x16x32_f16(false, af, false, bf0,
                                                      (short)0, acc0, false, false);
        acc1 = __builtin_amdgcn_wmma_f32_16x16x32_f16(false, af, false, bf1,
                                                      (short)0, acc1, false, false);
        acc2 = __builtin_amdgcn_wmma_f32_16x16x32_f16(false, af, false, bf2,
                                                      (short)0, acc2, false, false);
        acc3 = __builtin_amdgcn_wmma_f32_16x16x32_f16(false, af, false, bf3,
                                                      (short)0, acc3, false, false);
    }
    // ---- store: VGPR v holds M = Mbase + v (+8 for hi lanes), N = Nbase+16j+l15
    const int rbase = Mbase + (hi ? 8 : 0);
    float bias[8];
    #pragma unroll
    for (int v = 0; v < 8; ++v) bias[v] = biasTB[tb * C_ + rbase + v];
    #pragma unroll
    for (int v = 0; v < 8; ++v) {
        float* orow = out + ((size_t)tb * C_ + rbase + v) * HW_ + Nbase + l15;
        orow[0]  = acc0[v] + bias[v];
        orow[16] = acc1[v] + bias[v];
        orow[32] = acc2[v] + bias[v];
        orow[48] = acc3[v] + bias[v];
    }
}

// ---------------------------------------------------------------------------
extern "C" void kernel_launch(void* const* d_in, const int* in_sizes, int n_in,
                              void* d_out, int out_size, void* d_ws, size_t ws_size,
                              hipStream_t stream) {
    const float* x        = (const float*)d_in[0];
    const float* r1w      = (const float*)d_in[1];
    const float* r1b      = (const float*)d_in[2];
    const float* rlng     = (const float*)d_in[3];
    const float* rlnb     = (const float*)d_in[4];
    const float* r3w      = (const float*)d_in[5];
    const float* r3b      = (const float*)d_in[6];
    const float* rspw     = (const float*)d_in[7];
    const float* rspb     = (const float*)d_in[8];
    const float* rlnspg   = (const float*)d_in[9];
    const float* rlnspb   = (const float*)d_in[10];
    const float* r2w      = (const float*)d_in[11];
    const float* r2b      = (const float*)d_in[12];
    const float* routw    = (const float*)d_in[13];
    const float* routb    = (const float*)d_in[14];
    const float* eW       = (const float*)d_in[15];
    const float* eWb      = (const float*)d_in[16];
    const float* eM       = (const float*)d_in[17];
    const float* eMb      = (const float*)d_in[18];
    const float* ptW      = (const float*)d_in[19];
    const float* ptWb     = (const float*)d_in[20];
    const float* ptM      = (const float*)d_in[21];
    const float* ptMb     = (const float*)d_in[22];
    const float* pgw      = (const float*)d_in[23];
    const float* pgb      = (const float*)d_in[24];
    const float* bnallg   = (const float*)d_in[25];
    const float* bnallb   = (const float*)d_in[26];
    const float* bng      = (const float*)d_in[27];
    const float* bnb      = (const float*)d_in[28];
    float* out = (float*)d_out;

    char* ws = (char*)d_ws;
    float*    fmp    = (float*)   (ws + 0);        // 16*14*64 f = 57344 B
    float*    fsbuf  = (float*)   (ws + 57344);    // 4096 f    = 16384 B
    float*    route1 = (float*)   (ws + 73728);    // 240 f
    float*    prob   = (float*)   (ws + 74752);    // 16 f
    float*    MEb    = (float*)   (ws + 75008);    // 3840 f
    float*    PTb    = (float*)   (ws + 90368);    // 512 f
    float*    biasTB = (float*)   (ws + 92416);    // 4096 f
    float*    ME     = (float*)   (ws + 108800);   // 15*256*256 f = 3932160 B
    float*    PT     = (float*)   (ws + 4040960);  // 2*256*256 f  = 524288 B
    _Float16* Ah     = (_Float16*)(ws + 4565248);  // 16*256*256 h = 2097152 B
    _Float16* XhT    = (_Float16*)(ws + 6662400);  // 16*3136*256 h = 25690112 B

    k_router_feat<<<dim3(T_ * B_, 14), 256, 0, stream>>>(x, r1w, r1b, rlng, rlnb, fmp);
    k_router_sp<<<dim3(T_ * B_ * C_), 256, 0, stream>>>(x, rspw, fsbuf);
    k_router_gate<<<dim3(T_ * B_), 256, 0, stream>>>(fmp, fsbuf, rspb, rlnspg, rlnspb,
                                                     r3w, r3b, r2w, r2b, routw, routb,
                                                     route1, prob);
    k_rank_expand<<<dim3(C_, E_ + T_), 256, 0, stream>>>(eW, eWb, eM, eMb,
                                                         ptW, ptWb, ptM, ptMb,
                                                         ME, MEb, PT, PTb);
    k_combine<<<dim3(C_, T_ * B_), 256, 0, stream>>>(ME, MEb, PT, PTb, pgw, pgb,
                                                     bnallg, bnallb, bng, bnb,
                                                     route1, prob, Ah, biasTB);
    k_cvt_transpose<<<dim3(HW_ / 64, C_ / 64, T_ * B_), 256, 0, stream>>>(x, XhT);
    k_gemm_wmma<<<dim3(HW_ / 64, C_ / 128, T_ * B_), 256, 0, stream>>>(XhT, Ah, biasTB, out);
}
